// DPSNR_61546881351934
// MI455X (gfx1250) — matrix operand order repaired and checked by reading, not compiled, not run on hardware
//
#include <hip/hip_runtime.h>
#include <hip/hip_bf16.h>
#include <math.h>

// ---- problem dims ----
#define BB     4
#define SS     256
#define VV     50257
#define HD     256
#define FFD    512
#define NHD    4
#define DHD    64
#define LLD    2
#define POOLN  50000
#define LOOPSN 2
#define H4D    64
#define MROWS  (BB*SS)          // 1024
#define CHUNK  2048
#define CAND   (CHUNK + 64)
#define KMAX   512              // largest K any GEMM uses
#define KPAD   8                // LDS row padding (bf16 elems) to spread banks

// ---- types ----
typedef __bf16 bf16_t;
typedef __attribute__((ext_vector_type(16))) __bf16 v16bf;
typedef __attribute__((ext_vector_type(8)))  __bf16 v8bf;
typedef __attribute__((ext_vector_type(8)))  float  v8f;
typedef int v4i_t __attribute__((vector_size(16)));   // matches builtin param type

#define FLAG_ACC  1   // C += result (residual)
#define FLAG_GELU 2
#define FLAG_SIG  4

#if defined(__AMDGCN__) && __has_builtin(__builtin_amdgcn_global_load_async_to_lds_b128)
#define USE_ASYNC_LDS 1
#else
#define USE_ASYNC_LDS 0
#endif

__device__ __forceinline__ void wait_async_zero() {
#if __has_builtin(__builtin_amdgcn_s_wait_asynccnt)
  __builtin_amdgcn_s_wait_asynccnt(0);
#else
  asm volatile("s_wait_asynccnt 0" ::: "memory");
#endif
}

__device__ __forceinline__ bf16_t f2bf(float f) {
  union { float f; unsigned u; } a; a.f = f;
  unsigned r = a.u + 0x7FFFu + ((a.u >> 16) & 1u);   // round-to-nearest-even
  unsigned short h = (unsigned short)(r >> 16);
  return __builtin_bit_cast(bf16_t, h);
}
__device__ __forceinline__ float gelu_f(float x) {
  float x3 = x * x * x;
  return 0.5f * x * (1.f + tanhf(0.7978845608f * (x + 0.044715f * x3)));
}
__device__ __forceinline__ float sigm_f(float x) { return 1.f / (1.f + expf(-x)); }

// =====================================================================
// bf16 WMMA GEMM, register-blocked 4 N-tiles per wave, A strip staged in
// LDS (async global->LDS fill when available).
//   C[M,N](f32) = alpha * A[M,K] * Bt[N,K]^T (+bias)(+C)(act)
//   A row-major bf16 (lda), Bt stored [N,K] K-major bf16 (ldb), C f32 (ldc).
//   Optional batching: z -> (z/inner, z%inner) with outer/inner strides.
//   Fragment k order: contiguous k = 16*(lane/16) + s in A and B (wave32, 16x16x32 bf16).
// =====================================================================
__global__ __launch_bounds__(256) void gemm_bf16_wmma(
    const bf16_t* __restrict__ A, const bf16_t* __restrict__ Bt,
    float* __restrict__ C, bf16_t* __restrict__ Cbf,
    const float* __restrict__ bias,
    int M, int N, int K, int lda, int ldb, int ldc,
    int inner, long sAo, long sAi, long sBo, long sBi, long sCo, long sCi,
    float alpha, int flags)
{
  __shared__ bf16_t Atile[16 * (KMAX + KPAD)];

  const int wave = threadIdx.x >> 5;
  const int lane = threadIdx.x & 31;
  const int mtile = blockIdx.y;
  const int z = blockIdx.z;
  const long offA = (long)(z / inner) * sAo + (long)(z % inner) * sAi;
  const long offB = (long)(z / inner) * sBo + (long)(z % inner) * sBi;
  const long offC = (long)(z / inner) * sCo + (long)(z % inner) * sCi;
  const int ldk = K + KPAD;

  // ---- cooperative stage of the block's A strip [16 x K] into LDS ----
  {
    const bf16_t* Abase = A + offA + (long)(mtile * 16) * lda;
    const int kch = K >> 3;             // 16-byte chunks per row
    const int chunks = 16 * kch;
    for (int c = threadIdx.x; c < chunks; c += 256) {
      const int r = c / kch;
      const int kc = (c - r * kch) << 3;
      const bf16_t* gsrc = Abase + (long)r * lda + kc;
      bf16_t* ldst = &Atile[r * ldk + kc];
#if USE_ASYNC_LDS
      __builtin_amdgcn_global_load_async_to_lds_b128(
          (__attribute__((address_space(1))) v4i_t*)gsrc,
          (__attribute__((address_space(3))) v4i_t*)ldst, 0, 0);
#else
      *(v8bf*)ldst = *(const v8bf*)gsrc;
#endif
    }
#if USE_ASYNC_LDS
    wait_async_zero();
#endif
    __syncthreads();
  }

  const int ntiles = (N + 15) >> 4;
  const int tile0 = (blockIdx.x * 8 + wave) * 4;   // 4 N-tiles per wave
  if (tile0 >= ntiles) return;                     // wave-uniform (after barrier)

  const int l16 = lane & 15;
  const int kA0 = (lane < 16) ? 0 : 16;            // A: contiguous 16-K run per lane
  const int kB0 = (lane < 16) ? 0 : 16;            // B: contiguous 16-K run per lane

  int col[4];
  const bf16_t* Bcol[4];
#pragma unroll
  for (int t = 0; t < 4; ++t) {
    col[t] = (tile0 + t) * 16 + l16;
    int colc = (col[t] < N) ? col[t] : (N - 1);    // clamp: loads stay in-bounds, EXEC full
    Bcol[t] = Bt + offB + (long)colc * ldb;
  }

  v8f acc[4] = {{}, {}, {}, {}};
  const bf16_t* Alds = &Atile[l16 * ldk];

  for (int k = 0; k < K; k += 32) {
    v8bf a0 = *(const v8bf*)(Alds + k + kA0);      // ds_load_b128
    v8bf a1 = *(const v8bf*)(Alds + k + kA0 + 8);
    v16bf av;
#pragma unroll
    for (int i = 0; i < 8; ++i) { av[i] = a0[i]; av[i + 8] = a1[i]; }
#pragma unroll
    for (int t = 0; t < 4; ++t) {
      v16bf bv = *(const v16bf*)(Bcol[t] + k + kB0);
      __builtin_prefetch((const void*)(Bcol[t] + k + kB0 + 128), 0, 1); // global_prefetch_b8
      acc[t] = __builtin_amdgcn_wmma_f32_16x16x32_bf16(false, av, false, bv,
                                                       (short)0, acc[t], false, false);
    }
  }

  const int rbase = mtile * 16 + ((lane < 16) ? 0 : 8);  // C: VGPR r -> M = r (+8)
#pragma unroll
  for (int t = 0; t < 4; ++t) {
    if (col[t] < N) {
      const float bval = bias ? bias[col[t]] : 0.f;
#pragma unroll
      for (int r = 0; r < 8; ++r) {
        const long ci = offC + (long)(rbase + r) * ldc + col[t];
        float vv = acc[t][r] * alpha + bval;
        if (flags & FLAG_ACC)  vv += C[ci];
        if (flags & FLAG_GELU) vv = gelu_f(vv);
        else if (flags & FLAG_SIG) vv = sigm_f(vv);
        C[ci] = vv;
        if (Cbf) Cbf[ci] = f2bf(vv);
      }
    }
  }
}

// ---------------- small support kernels ----------------

__global__ void cvt_bf16_kernel(const float* __restrict__ x, bf16_t* __restrict__ y, int n) {
  int i = blockIdx.x * 256 + threadIdx.x;
  if (i < n) y[i] = f2bf(x[i]);
}

// src [K,N] f32 row-major -> dst [N,K] bf16 K-major
__global__ void tconv_bf16_kernel(const float* __restrict__ src, bf16_t* __restrict__ dst,
                                  int K, int N) {
  int i = blockIdx.x * 256 + threadIdx.x;
  if (i >= K * N) return;
  int n = i / K, k = i % K;
  dst[i] = f2bf(src[(long)k * N + n]);
}

// v f32 [B,S,NH,DH] -> vT bf16 [B,NH,DH,S]
__global__ void transpose_v_kernel(const float* __restrict__ v, bf16_t* __restrict__ vT) {
  int i = blockIdx.x * 256 + threadIdx.x;   // 2^18 total
  int s = i & 255;
  int d = (i >> 8) & 63;
  int h = (i >> 14) & 3;
  int b = i >> 16;
  vT[i] = f2bf(v[(((b * SS + s) * NHD + h) * DHD) + d]);
}

__global__ void embed_kernel(const int* __restrict__ tokens, const float* __restrict__ tokemb,
                             const float* __restrict__ posemb, float* __restrict__ h,
                             float* __restrict__ cum, float* __restrict__ halted) {
  int row = blockIdx.x, d = threadIdx.x;
  int tok = tokens[row];
  h[row * HD + d] = tokemb[(long)tok * HD + d] + posemb[(row & (SS - 1)) * HD + d];
  if (d == 0) { cum[row] = 0.f; halted[row] = 0.f; }
}

__global__ void loopemb_kernel(const float* __restrict__ hidden,
                               const float* __restrict__ loop_table, int it,
                               float* __restrict__ x) {
  int row = blockIdx.x, d = threadIdx.x;
  int j = d >> 1;
  float div = expf(-(logf(10000.f) * (float)(2 * j)) / (float)HD);
  float pe = (d & 1) ? cosf((float)it * div) : sinf((float)it * div);
  x[row * HD + d] = hidden[row * HD + d] + loop_table[it * HD + d] + pe;
}

__global__ void ln_kernel(const float* __restrict__ x, const float* __restrict__ s,
                          const float* __restrict__ b, float* __restrict__ y,
                          bf16_t* __restrict__ ybf) {
  __shared__ float red[256];
  int row = blockIdx.x, t = threadIdx.x;
  float v = x[row * HD + t];
  red[t] = v; __syncthreads();
  for (int st = 128; st > 0; st >>= 1) { if (t < st) red[t] += red[t + st]; __syncthreads(); }
  float m = red[0] * (1.f / HD); __syncthreads();
  float d = v - m;
  red[t] = d * d; __syncthreads();
  for (int st = 128; st > 0; st >>= 1) { if (t < st) red[t] += red[t + st]; __syncthreads(); }
  float var = red[0] * (1.f / HD);
  float o = d * rsqrtf(var + 1e-6f) * s[t] + b[t];
  y[row * HD + t] = o;
  if (ybf) ybf[row * HD + t] = f2bf(o);
}

// att f32 [B,NH,S,S] -> causal softmax -> probs bf16
__global__ void softmax_kernel(const float* __restrict__ att, bf16_t* __restrict__ probs) {
  __shared__ float red[256];
  int row = blockIdx.x, t = threadIdx.x;
  int q = row & (SS - 1);
  const float* a = att + (long)row * SS;
  float x = (t <= q) ? a[t] : -1e9f;
  red[t] = x; __syncthreads();
  for (int st = 128; st > 0; st >>= 1) { if (t < st) red[t] = fmaxf(red[t], red[t + st]); __syncthreads(); }
  float m = red[0]; __syncthreads();
  float e = expf(x - m);
  red[t] = e; __syncthreads();
  for (int st = 128; st > 0; st >>= 1) { if (t < st) red[t] += red[t + st]; __syncthreads(); }
  probs[(long)row * SS + t] = f2bf(e / red[0]);
}

__global__ void kval_kernel(const float* __restrict__ proc, const float* __restrict__ kW,
                            const float* __restrict__ kb, int* __restrict__ kvals) {
  __shared__ float red[256];
  int row = blockIdx.x, t = threadIdx.x;
  red[t] = proc[row * HD + t] * kW[t]; __syncthreads();
  for (int st = 128; st > 0; st >>= 1) { if (t < st) red[t] += red[t + st]; __syncthreads(); }
  if (t == 0) {
    float sg = sigm_f(red[0] + kb[0]);
    kvals[row] = (int)floorf(16.f + sg * 48.f);   // [MINK, MAXK)
  }
}

// exact running top-64 merge over one score chunk (descending, like top_k)
__global__ __launch_bounds__(256) void topk_merge_kernel(
    const float* __restrict__ scores, int chunkStart, int chunkN,
    float* __restrict__ topv, int* __restrict__ topi, int firstChunk) {
  __shared__ float cv[CAND];
  __shared__ int   ci[CAND];
  __shared__ float rv[256];
  __shared__ int   rp[256];
  int row = blockIdx.x, t = threadIdx.x;
  const float* srow = scores + (long)row * CHUNK;
  for (int j = t; j < CHUNK; j += 256) {
    cv[j] = (j < chunkN) ? srow[j] : -1e30f;
    ci[j] = chunkStart + j;
  }
  for (int j = t; j < 64; j += 256) {
    cv[CHUNK + j] = firstChunk ? -1e30f : topv[row * 64 + j];
    ci[CHUNK + j] = firstChunk ? 0      : topi[row * 64 + j];
  }
  __syncthreads();
  for (int it = 0; it < 64; ++it) {
    float bv = -1e30f; int bp = 0;
    for (int j = t; j < CAND; j += 256) if (cv[j] > bv) { bv = cv[j]; bp = j; }
    rv[t] = bv; rp[t] = bp; __syncthreads();
    for (int st = 128; st > 0; st >>= 1) {
      if (t < st && rv[t + st] > rv[t]) { rv[t] = rv[t + st]; rp[t] = rp[t + st]; }
      __syncthreads();
    }
    if (t == 0) {
      int p = rp[0];
      topv[row * 64 + it] = rv[0];
      topi[row * 64 + it] = ci[p];
      cv[p] = -1e30f;
    }
    __syncthreads();
  }
}

__global__ void context_kernel(const float* __restrict__ pool, const int* __restrict__ topi,
                               const int* __restrict__ kvals, float* __restrict__ ctx) {
  int row = blockIdx.x, d = threadIdx.x;
  int kv = kvals[row];
  float s = 0.f;
  for (int j = 0; j < kv; ++j) s += pool[(long)topi[row * 64 + j] * HD + d];
  ctx[row * HD + d] = s / ((float)kv + 1e-6f);
}

// out bf16 [rows, 2H] = concat(X[rows,H], Y[rows,H])
__global__ void concat_bf_kernel(const float* __restrict__ X, const float* __restrict__ Y,
                                 bf16_t* __restrict__ out) {
  int row = blockIdx.x, d = threadIdx.x;
  out[row * (2 * HD) + d]      = f2bf(X[row * HD + d]);
  out[row * (2 * HD) + HD + d] = f2bf(Y[row * HD + d]);
}

__global__ void gate_mix_kernel(const float* __restrict__ gate, const float* __restrict__ tb,
                                const float* __restrict__ hidden, float* __restrict__ pre, int n) {
  int i = blockIdx.x * 256 + threadIdx.x;
  if (i < n) { float g = gate[i]; pre[i] = g * tb[i] + (1.f - g) * hidden[i]; }
}

__global__ void act_update_kernel(const float* __restrict__ newacc, const float* __restrict__ hp,
                                  float* __restrict__ hidden, float* __restrict__ cum,
                                  float* __restrict__ halted) {
  int row = blockIdx.x, d = threadIdx.x;
  float hal = halted[row];
  float still = 1.f - hal;
  float h  = hidden[row * HD + d];
  float na = newacc[row * HD + d];
  __syncthreads();
  hidden[row * HD + d] = still * na + (1.f - still) * h;
  if (d == 0) {
    float c = cum[row] + hp[row] * still;
    cum[row] = c;
    halted[row] = fmaxf(hal, (c >= 1.f) ? 1.f : 0.f);
  }
}

// ---------------- host side ----------------

static inline void launch_gemm(hipStream_t st, const bf16_t* A, const bf16_t* Bt,
                               float* C, bf16_t* Cbf, const float* bias,
                               int M, int N, int K, int lda, int ldb, int ldc,
                               float alpha, int flags,
                               int batches = 1, int inner = 1,
                               long sAo = 0, long sAi = 0, long sBo = 0, long sBi = 0,
                               long sCo = 0, long sCi = 0) {
  int ntiles = (N + 15) / 16;
  dim3 grid((unsigned)((ntiles + 31) / 32), (unsigned)(M / 16), (unsigned)batches);
  gemm_bf16_wmma<<<grid, 256, 0, st>>>(A, Bt, C, Cbf, bias, M, N, K, lda, ldb, ldc,
                                       inner, sAo, sAi, sBo, sBi, sCo, sCi, alpha, flags);
}

extern "C" void kernel_launch(void* const* d_in, const int* in_sizes, int n_in,
                              void* d_out, int out_size, void* d_ws, size_t ws_size,
                              hipStream_t stream) {
  (void)in_sizes; (void)n_in; (void)out_size; (void)ws_size;

  const int*   tokens   = (const int*)  d_in[0];
  const float* tok_emb  = (const float*)d_in[1];
  const float* pos_emb  = (const float*)d_in[2];
  const float* ln1_s    = (const float*)d_in[3];
  const float* ln1_b    = (const float*)d_in[4];
  const float* Wq       = (const float*)d_in[5];
  const float* bq       = (const float*)d_in[6];
  const float* Wk       = (const float*)d_in[7];
  const float* bk       = (const float*)d_in[8];
  const float* Wv       = (const float*)d_in[9];
  const float* bv       = (const float*)d_in[10];
  const float* Wo       = (const float*)d_in[11];
  const float* bo       = (const float*)d_in[12];
  const float* ln2_s    = (const float*)d_in[13];
  const float* ln2_b    = (const float*)d_in[14];
  const float* W1       = (const float*)d_in[15];
  const float* b1       = (const float*)d_in[16];
  const float* W2       = (const float*)d_in[17];
  const float* b2       = (const float*)d_in[18];
  const float* pool     = (const float*)d_in[19];
  const float* halt_W1  = (const float*)d_in[20];
  const float* halt_b1  = (const float*)d_in[21];
  const float* halt_W2  = (const float*)d_in[22];
  const float* halt_b2  = (const float*)d_in[23];
  const float* acc_gW   = (const float*)d_in[24];
  const float* acc_gb   = (const float*)d_in[25];
  const float* acc_tW   = (const float*)d_in[26];
  const float* acc_tb   = (const float*)d_in[27];
  const float* acc_ln_s = (const float*)d_in[28];
  const float* acc_ln_b = (const float*)d_in[29];
  const float* loop_tab = (const float*)d_in[30];
  const float* int_W1   = (const float*)d_in[31];
  const float* int_b1   = (const float*)d_in[32];
  const float* int_W2   = (const float*)d_in[33];
  const float* int_b2   = (const float*)d_in[34];
  const float* k_W      = (const float*)d_in[35];
  const float* k_b      = (const float*)d_in[36];
  const float* lm_W     = (const float*)d_in[37];
  const float* lm_b     = (const float*)d_in[38];

  // ---- bump-allocate workspace (256B aligned) ----
  char* base = (char*)d_ws;
  size_t off = 0;
  auto alloc = [&](size_t bytes) -> char* {
    char* p = base + off;
    off = (off + bytes + 255) & ~(size_t)255;
    return p;
  };
  // bf16 [N,K] weights
  bf16_t* WqT   = (bf16_t*)alloc((size_t)LLD * HD * HD * 2);
  bf16_t* WkT   = (bf16_t*)alloc((size_t)LLD * HD * HD * 2);
  bf16_t* WvT   = (bf16_t*)alloc((size_t)LLD * HD * HD * 2);
  bf16_t* WoT   = (bf16_t*)alloc((size_t)LLD * HD * HD * 2);
  bf16_t* W1T   = (bf16_t*)alloc((size_t)LLD * FFD * HD * 2);
  bf16_t* W2T   = (bf16_t*)alloc((size_t)LLD * HD * FFD * 2);
  bf16_t* iW1T  = (bf16_t*)alloc((size_t)(2 * HD) * (2 * HD) * 2);
  bf16_t* iW2T  = (bf16_t*)alloc((size_t)HD * (2 * HD) * 2);
  bf16_t* gWT   = (bf16_t*)alloc((size_t)HD * (2 * HD) * 2);
  bf16_t* tWT   = (bf16_t*)alloc((size_t)HD * HD * 2);
  bf16_t* hW1T  = (bf16_t*)alloc((size_t)H4D * HD * 2);
  bf16_t* hW2T  = (bf16_t*)alloc((size_t)1 * H4D * 2);
  bf16_t* lmWT  = (bf16_t*)alloc((size_t)VV * HD * 2);
  bf16_t* poolB = (bf16_t*)alloc((size_t)POOLN * HD * 2);
  // activations
  float*  hid   = (float*) alloc((size_t)MROWS * HD * 4);
  float*  x     = (float*) alloc((size_t)MROWS * HD * 4);
  float*  y     = (float*) alloc((size_t)MROWS * HD * 4);
  bf16_t* ybf   = (bf16_t*)alloc((size_t)MROWS * HD * 2);
  float*  q     = (float*) alloc((size_t)MROWS * HD * 4);
  float*  k     = (float*) alloc((size_t)MROWS * HD * 4);
  float*  v     = (float*) alloc((size_t)MROWS * HD * 4);
  bf16_t* qbf   = (bf16_t*)alloc((size_t)MROWS * HD * 2);
  bf16_t* kbf   = (bf16_t*)alloc((size_t)MROWS * HD * 2);
  float*  att   = (float*) alloc((size_t)BB * NHD * SS * SS * 4);
  bf16_t* attp  = (bf16_t*)alloc((size_t)BB * NHD * SS * SS * 2);
  bf16_t* vT    = (bf16_t*)alloc((size_t)BB * NHD * DHD * SS * 2);
  float*  o     = (float*) alloc((size_t)MROWS * HD * 4);
  bf16_t* obf   = (bf16_t*)alloc((size_t)MROWS * HD * 2);
  float*  mid   = (float*) alloc((size_t)MROWS * FFD * 4);
  bf16_t* midbf = (bf16_t*)alloc((size_t)MROWS * FFD * 2);
  bf16_t* prcbf = (bf16_t*)alloc((size_t)MROWS * HD * 2);
  float*  scor  = (float*) alloc((size_t)MROWS * CHUNK * 4);
  float*  topv  = (float*) alloc((size_t)MROWS * 64 * 4);
  int*    topi  = (int*)   alloc((size_t)MROWS * 64 * 4);
  int*    kvals = (int*)   alloc((size_t)MROWS * 4);
  float*  ctx   = (float*) alloc((size_t)MROWS * HD * 4);
  bf16_t* ccbf  = (bf16_t*)alloc((size_t)MROWS * 2 * HD * 2);
  float*  integ = (float*) alloc((size_t)MROWS * HD * 4);
  bf16_t* intbf = (bf16_t*)alloc((size_t)MROWS * HD * 2);
  bf16_t* cmbbf = (bf16_t*)alloc((size_t)MROWS * 2 * HD * 2);
  float*  gate  = (float*) alloc((size_t)MROWS * HD * 4);
  float*  tbuf  = (float*) alloc((size_t)MROWS * HD * 4);
  float*  pre   = (float*) alloc((size_t)MROWS * HD * 4);
  float*  nacc  = (float*) alloc((size_t)MROWS * HD * 4);
  bf16_t* naccb = (bf16_t*)alloc((size_t)MROWS * HD * 2);
  float*  g1    = (float*) alloc((size_t)MROWS * H4D * 4);
  bf16_t* g1bf  = (bf16_t*)alloc((size_t)MROWS * H4D * 2);
  float*  hp    = (float*) alloc((size_t)MROWS * 4);
  float*  cum   = (float*) alloc((size_t)MROWS * 4);
  float*  hltd  = (float*) alloc((size_t)MROWS * 4);
  bf16_t* hidbf = (bf16_t*)alloc((size_t)MROWS * HD * 2);

  // ---- prep: convert/transpose all weights to bf16 [N,K] ----
  auto tconv = [&](const float* src, bf16_t* dst, int Kd, int Nd) {
    int n = Kd * Nd;
    tconv_bf16_kernel<<<(n + 255) / 256, 256, 0, stream>>>(src, dst, Kd, Nd);
  };
  for (int l = 0; l < LLD; ++l) {
    tconv(Wq + (size_t)l * HD * HD, WqT + (size_t)l * HD * HD, HD, HD);
    tconv(Wk + (size_t)l * HD * HD, WkT + (size_t)l * HD * HD, HD, HD);
    tconv(Wv + (size_t)l * HD * HD, WvT + (size_t)l * HD * HD, HD, HD);
    tconv(Wo + (size_t)l * HD * HD, WoT + (size_t)l * HD * HD, HD, HD);
    tconv(W1 + (size_t)l * HD * FFD, W1T + (size_t)l * FFD * HD, HD, FFD);
    tconv(W2 + (size_t)l * FFD * HD, W2T + (size_t)l * HD * FFD, FFD, HD);
  }
  tconv(int_W1,  iW1T, 2 * HD, 2 * HD);
  tconv(int_W2,  iW2T, 2 * HD, HD);
  tconv(acc_gW,  gWT,  2 * HD, HD);
  tconv(acc_tW,  tWT,  HD, HD);
  tconv(halt_W1, hW1T, HD, H4D);
  tconv(halt_W2, hW2T, H4D, 1);
  tconv(lm_W,    lmWT, HD, VV);
  cvt_bf16_kernel<<<(POOLN * HD + 255) / 256, 256, 0, stream>>>(pool, poolB, POOLN * HD);

  // ---- embedding + ACT state init ----
  embed_kernel<<<MROWS, HD, 0, stream>>>(tokens, tok_emb, pos_emb, hid, cum, hltd);

  const float scale = 1.f / sqrtf((float)DHD);

  for (int it = 0; it < LOOPSN; ++it) {
    // x = hidden + loop_table[it] + sinusoidal PE
    loopemb_kernel<<<MROWS, HD, 0, stream>>>(hid, loop_tab, it, x);

    // ---- controller: L transformer layers ----
    for (int l = 0; l < LLD; ++l) {
      ln_kernel<<<MROWS, HD, 0, stream>>>(x, ln1_s + l * HD, ln1_b + l * HD, y, ybf);
      launch_gemm(stream, ybf, WqT + (size_t)l * HD * HD, q, qbf, bq + l * HD,
                  MROWS, HD, HD, HD, HD, HD, 1.f, 0);
      launch_gemm(stream, ybf, WkT + (size_t)l * HD * HD, k, kbf, bk + l * HD,
                  MROWS, HD, HD, HD, HD, HD, 1.f, 0);
      launch_gemm(stream, ybf, WvT + (size_t)l * HD * HD, v, nullptr, bv + l * HD,
                  MROWS, HD, HD, HD, HD, HD, 1.f, 0);
      // att[b,h] = scale * q[b,:,h,:] @ k[b,:,h,:]^T  (batched over B*NH)
      launch_gemm(stream, qbf, kbf, att, nullptr, nullptr,
                  SS, SS, DHD, HD, HD, SS, scale, 0,
                  BB * NHD, NHD,
                  (long)SS * HD, (long)DHD, (long)SS * HD, (long)DHD,
                  (long)NHD * SS * SS, (long)SS * SS);
      softmax_kernel<<<BB * NHD * SS, SS, 0, stream>>>(att, attp);
      transpose_v_kernel<<<(BB * NHD * DHD * SS) / 256, 256, 0, stream>>>(v, vT);
      // o[b,:,h,:] = probs[b,h] @ v[b,:,h,:]
      launch_gemm(stream, attp, vT, o, obf, nullptr,
                  SS, DHD, SS, SS, SS, HD, 1.f, 0,
                  BB * NHD, NHD,
                  (long)NHD * SS * SS, (long)SS * SS,
                  (long)NHD * DHD * SS, (long)DHD * SS,
                  (long)SS * HD, (long)DHD);
      // x += o @ Wo + bo
      launch_gemm(stream, obf, WoT + (size_t)l * HD * HD, x, nullptr, bo + l * HD,
                  MROWS, HD, HD, HD, HD, HD, 1.f, FLAG_ACC);
      // MLP
      ln_kernel<<<MROWS, HD, 0, stream>>>(x, ln2_s + l * HD, ln2_b + l * HD, y, ybf);
      launch_gemm(stream, ybf, W1T + (size_t)l * FFD * HD, mid, midbf, b1 + l * FFD,
                  MROWS, FFD, HD, HD, HD, FFD, 1.f, FLAG_GELU);
      launch_gemm(stream, midbf, W2T + (size_t)l * HD * FFD, x, nullptr, b2 + l * HD,
                  MROWS, HD, FFD, FFD, FFD, HD, 1.f, FLAG_ACC);
    }
    // proc = x
    cvt_bf16_kernel<<<(MROWS * HD + 255) / 256, 256, 0, stream>>>(x, prcbf, MROWS * HD);
    kval_kernel<<<MROWS, HD, 0, stream>>>(x, k_W, k_b, kvals);

    // ---- massive pool: chunked WMMA scoring + exact running top-64 ----
    const int nchunks = (POOLN + CHUNK - 1) / CHUNK;
    for (int c = 0; c < nchunks; ++c) {
      int start = c * CHUNK;
      int cn = (POOLN - start < CHUNK) ? (POOLN - start) : CHUNK;
      launch_gemm(stream, prcbf, poolB + (size_t)start * HD, scor, nullptr, nullptr,
                  MROWS, cn, HD, HD, HD, CHUNK, 1.f, 0);
      topk_merge_kernel<<<MROWS, 256, 0, stream>>>(scor, start, cn, topv, topi, c == 0);
    }
    context_kernel<<<MROWS, HD, 0, stream>>>(pool, topi, kvals, ctx);

    // ---- integrator ----
    concat_bf_kernel<<<MROWS, HD, 0, stream>>>(x, ctx, ccbf);
    launch_gemm(stream, ccbf, iW1T, mid, midbf, int_b1,
                MROWS, 2 * HD, 2 * HD, 2 * HD, 2 * HD, 2 * HD, 1.f, FLAG_GELU);
    launch_gemm(stream, midbf, iW2T, integ, intbf, int_b2,
                MROWS, HD, 2 * HD, 2 * HD, 2 * HD, HD, 1.f, 0);

    // ---- gated state accumulator + LN ----
    concat_bf_kernel<<<MROWS, HD, 0, stream>>>(integ, hid, cmbbf);
    launch_gemm(stream, cmbbf, gWT, gate, nullptr, acc_gb,
                MROWS, HD, 2 * HD, 2 * HD, 2 * HD, HD, 1.f, FLAG_SIG);
    launch_gemm(stream, intbf, tWT, tbuf, nullptr, acc_tb,
                MROWS, HD, HD, HD, HD, HD, 1.f, 0);
    gate_mix_kernel<<<(MROWS * HD) / 256, 256, 0, stream>>>(gate, tbuf, hid, pre, MROWS * HD);
    ln_kernel<<<MROWS, HD, 0, stream>>>(pre, acc_ln_s, acc_ln_b, nacc, naccb);

    // ---- ACT halting ----
    launch_gemm(stream, naccb, hW1T, g1, g1bf, halt_b1,
                MROWS, H4D, HD, HD, HD, H4D, 1.f, FLAG_GELU);
    launch_gemm(stream, g1bf, hW2T, hp, nullptr, halt_b2,
                MROWS, 1, H4D, H4D, H4D, 1, 1.f, FLAG_SIG);
    act_update_kernel<<<MROWS, HD, 0, stream>>>(nacc, hp, hid, cum, hltd);
  }

  // ---- LM head: logits = hidden @ lm_W + lm_b  -> d_out [B,S,V] f32 ----
  cvt_bf16_kernel<<<(MROWS * HD + 255) / 256, 256, 0, stream>>>(hid, hidbf, MROWS * HD);
  launch_gemm(stream, hidbf, lmWT, (float*)d_out, nullptr, lm_b,
              MROWS, VV, HD, HD, HD, VV, 1.f, 0);
}
